// MultiHeadAttention_6356551598312
// MI455X (gfx1250) — matrix operand order repaired and checked by reading, compiled, not run on hardware
//
#include <hip/hip_runtime.h>
#include <hip/hip_bf16.h>

// ---------------------------------------------------------------------------
// MultiHeadAttention for MI455X (gfx1250): all GEMMs via v_wmma_f32_16x16x32_bf16
//   1) Q/K/V projections: f32 in -> bf16 out (bias fused)
//   2) logits = Q K^T (per batch), f32 out
//   3) softmax rows + quirk scale folded into bf16 probs
//   4) out = P V, f32 out
// B=8, S=2048, D=1024
// Pipelined: double-buffered LDS ping-pong, register prefetch of next tile,
// global_prefetch of tile-after-next, b128 staging loads, native __bf16
// casts so the backend picks the hardware f32->bf16 convert.
// ---------------------------------------------------------------------------

typedef __attribute__((ext_vector_type(16))) __bf16 v16bf;
typedef __attribute__((ext_vector_type(2)))  __bf16 v2bf;
typedef __attribute__((ext_vector_type(8)))  float  v8f;

union FragB16 {
    unsigned int u[8];
    v16bf v;
};

// convert+pack two floats into two bf16 (backend selects HW cvt when present)
__device__ __forceinline__ unsigned int pack2_bf16(float lo, float hi) {
    v2bf pk;
    pk[0] = (__bf16)lo;
    pk[1] = (__bf16)hi;
    return __builtin_bit_cast(unsigned int, pk);
}

__device__ __forceinline__ unsigned short f32_to_bf16_one(float f) {
    __bf16 b = (__bf16)f;
    return __builtin_bit_cast(unsigned short, b);
}

__device__ __forceinline__ v8f zero8() {
    v8f z;
#pragma unroll
    for (int i = 0; i < 8; ++i) z[i] = 0.0f;
    return z;
}

// ---------------------------------------------------------------------------
// Templated WMMA GEMM:  C[M,N] = A[M,K] * B(+bias)
//   A_F32:   1 -> A is float (convert to bf16 while staging), 0 -> A is bf16 bits
//   B_MODE:  0 -> B is f32 [K,N] (weights), 1 -> B is bf16 [N,K] (K^T access),
//            2 -> B is bf16 [K,N] (V)
//   HAS_BIAS: add f32 bias[n]
//   OUT_F32: 1 -> store f32, 0 -> store bf16 bits
// Block tile 128x128, BK=64, 256 threads = 8 wave32 waves (4 x 2),
// each wave computes a 32x64 tile = 2x4 WMMA 16x16 accumulators.
// ---------------------------------------------------------------------------
template <int BM, int BN, int BK, int A_F32, int B_MODE, int HAS_BIAS, int OUT_F32>
__global__ __launch_bounds__(256) void wmma_gemm_kernel(
    const void* __restrict__ Av, const void* __restrict__ Bv,
    const float* __restrict__ bias, void* __restrict__ Cv,
    int M, int N, int K,
    long long strideA, long long strideB, long long strideC)
{
    constexpr int PAD = 8;                         // row stride 72 ushorts = 144 B
    __shared__ __align__(16) unsigned short Asm[2][BM][BK + PAD];  // [m][k]
    __shared__ __align__(16) unsigned short Bsm[2][BN][BK + PAD];  // [n][k]

    const int tid    = threadIdx.x;
    const int wave   = tid >> 5;
    const int lane   = tid & 31;
    const int halfId = lane >> 4;
    const int r      = lane & 15;

    const int wm = wave & 3;              // 4 waves along M
    const int wn = wave >> 2;             // 2 waves along N
    const int rowOff = wm * 32;
    const int colOff = wn * 64;

    const int m0 = blockIdx.y * BM;
    const int n0 = blockIdx.x * BN;
    const long long z = blockIdx.z;

    const float*          Af = (const float*)Av + z * strideA;
    const unsigned short* Ab = (const unsigned short*)Av + z * strideA;
    const float*          Bf = (const float*)Bv + z * strideB;
    const unsigned short* Bb = (const unsigned short*)Bv + z * strideB;

    // ---- per-thread staging coordinates ----
    const int sa_m = tid >> 1;            // A: row = tid/2, k = (tid&1)*32
    const int sa_k = (tid & 1) * 32;
    int sb_n, sb_k;
    if constexpr (B_MODE == 0) { sb_n = (tid & 31) * 4; sb_k = (tid >> 5) * 8; }
    else if constexpr (B_MODE == 1) { sb_n = tid >> 1;  sb_k = (tid & 1) * 32; }
    else { sb_n = (tid & 15) * 8; sb_k = (tid >> 4) * 4; }

    // staging registers
    float4 aF[8]; uint4 aH[4];
    float4 bF[8]; uint4 bH[4];

    auto globalLoad = [&](int k0) {
        if constexpr (A_F32) {
            const float* src = Af + (size_t)(m0 + sa_m) * (size_t)K + (size_t)(k0 + sa_k);
#pragma unroll
            for (int j = 0; j < 8; ++j) aF[j] = *(const float4*)(src + 4 * j);
        } else {
            const unsigned short* src = Ab + (size_t)(m0 + sa_m) * (size_t)K + (size_t)(k0 + sa_k);
#pragma unroll
            for (int j = 0; j < 4; ++j) aH[j] = *(const uint4*)(src + 8 * j);
        }
        if constexpr (B_MODE == 0) {          // f32 [K,N]
            const float* src = Bf + (size_t)(k0 + sb_k) * (size_t)N + (size_t)(n0 + sb_n);
#pragma unroll
            for (int j = 0; j < 8; ++j) bF[j] = *(const float4*)(src + (size_t)j * N);
        } else if constexpr (B_MODE == 1) {   // bf16 [N,K]
            const unsigned short* src = Bb + (size_t)(n0 + sb_n) * (size_t)K + (size_t)(k0 + sb_k);
#pragma unroll
            for (int j = 0; j < 4; ++j) bH[j] = *(const uint4*)(src + 8 * j);
        } else {                              // bf16 [K,N]
            const unsigned short* src = Bb + (size_t)(k0 + sb_k) * (size_t)N + (size_t)(n0 + sb_n);
#pragma unroll
            for (int j = 0; j < 4; ++j) bH[j] = *(const uint4*)(src + (size_t)j * N);
        }
    };

    auto ldsStore = [&](int buf) {
        if constexpr (A_F32) {
#pragma unroll
            for (int j = 0; j < 8; ++j) {
                uint2 pk;
                pk.x = pack2_bf16(aF[j].x, aF[j].y);
                pk.y = pack2_bf16(aF[j].z, aF[j].w);
                *(uint2*)&Asm[buf][sa_m][sa_k + 4 * j] = pk;
            }
        } else {
#pragma unroll
            for (int j = 0; j < 4; ++j)
                *(uint4*)&Asm[buf][sa_m][sa_k + 8 * j] = aH[j];
        }
        if constexpr (B_MODE == 0) {
            // pack adjacent-k pairs per destination row: 16 b32 stores
#pragma unroll
            for (int c = 0; c < 4; ++c) {
#pragma unroll
                for (int j = 0; j < 8; j += 2) {
                    const float lo = ((const float*)&bF[j])[c];
                    const float hi = ((const float*)&bF[j + 1])[c];
                    *(unsigned int*)&Bsm[buf][sb_n + c][sb_k + j] = pack2_bf16(lo, hi);
                }
            }
        } else if constexpr (B_MODE == 1) {
#pragma unroll
            for (int j = 0; j < 4; ++j)
                *(uint4*)&Bsm[buf][sb_n][sb_k + 8 * j] = bH[j];
        } else {
#pragma unroll
            for (int j = 0; j < 4; ++j) {
                const unsigned int* pw = (const unsigned int*)&bH[j];
#pragma unroll
                for (int c = 0; c < 8; ++c)
                    Bsm[buf][sb_n + c][sb_k + j] =
                        (unsigned short)(pw[c >> 1] >> ((c & 1) * 16));
            }
        }
    };

    v8f acc[2][4];
#pragma unroll
    for (int i = 0; i < 2; ++i)
#pragma unroll
        for (int j = 0; j < 4; ++j) acc[i][j] = zero8();

    const int nIter = K / BK;
    globalLoad(0);
    ldsStore(0);
    __syncthreads();

    for (int it = 0; it < nIter; ++it) {
        const int cur = it & 1;

        if (it + 1 < nIter) {
            globalLoad((it + 1) * BK);
            if (it + 2 < nIter) {   // pull tile-after-next toward L2 (global_prefetch_b8)
                const int kp = (it + 2) * BK;
                if constexpr (A_F32)
                    __builtin_prefetch(Af + (size_t)(m0 + sa_m) * (size_t)K + (size_t)(kp + sa_k), 0, 3);
                else
                    __builtin_prefetch(Ab + (size_t)(m0 + sa_m) * (size_t)K + (size_t)(kp + sa_k), 0, 3);
                if constexpr (B_MODE == 0)
                    __builtin_prefetch(Bf + (size_t)(kp + sb_k) * (size_t)N + (size_t)(n0 + sb_n), 0, 3);
                else if constexpr (B_MODE == 1)
                    __builtin_prefetch(Bb + (size_t)(n0 + sb_n) * (size_t)K + (size_t)(kp + sb_k), 0, 3);
                else
                    __builtin_prefetch(Bb + (size_t)(kp + sb_k) * (size_t)N + (size_t)(n0 + sb_n), 0, 3);
            }
        }

        // ---- compute on current buffer while next tile's loads are in flight ----
#pragma unroll
        for (int ks = 0; ks < BK; ks += 32) {
            FragB16 fb[4];
#pragma unroll
            for (int nt = 0; nt < 4; ++nt) {
                const int n  = colOff + nt * 16 + r;
                const int kb = ks + halfId * 16;
#pragma unroll
                for (int p = 0; p < 8; ++p)
                    fb[nt].u[p] = *(const unsigned int*)&Bsm[cur][n][kb + 2 * p];
            }
#pragma unroll
            for (int mt = 0; mt < 2; ++mt) {
                const int m  = rowOff + mt * 16 + r;
                const int kb = ks + halfId * 16;
                FragB16 fa;
#pragma unroll
                for (int p = 0; p < 8; ++p) {
                    const int kk = kb + 2 * p;
                    fa.u[p] = *(const unsigned int*)&Asm[cur][m][kk];
                }
#pragma unroll
                for (int nt = 0; nt < 4; ++nt) {
                    acc[mt][nt] = __builtin_amdgcn_wmma_f32_16x16x32_bf16(
                        false, fa.v, false, fb[nt].v,
                        (short)0, acc[mt][nt], false, false);
                }
            }
        }

        if (it + 1 < nIter) ldsStore(cur ^ 1);
        __syncthreads();
    }

    // ---- epilogue: C/D layout -> lane r = col, VGPR v + halfId*8 = row ----
    float*          Cf = (float*)Cv + z * strideC;
    unsigned short* Cb = (unsigned short*)Cv + z * strideC;
#pragma unroll
    for (int mt = 0; mt < 2; ++mt) {
#pragma unroll
        for (int nt = 0; nt < 4; ++nt) {
            const int gn = n0 + colOff + nt * 16 + r;
            const float bv = HAS_BIAS ? bias[gn] : 0.0f;
#pragma unroll
            for (int v = 0; v < 8; ++v) {
                const int gm = m0 + rowOff + mt * 16 + halfId * 8 + v;
                const float val = acc[mt][nt][v] + bv;
                const size_t cidx = (size_t)gm * (size_t)N + (size_t)gn;
                if (OUT_F32) Cf[cidx] = val;
                else         Cb[cidx] = f32_to_bf16_one(val);
            }
        }
    }
}

// ---------------------------------------------------------------------------
// Row softmax with the reference quirk: P = softmax(logits) * scale, bf16 out.
// One 256-thread block per row of S=2048; each thread owns 8 contiguous cols
// (b128 loads, packed cvt, b128 store).
// ---------------------------------------------------------------------------
__global__ __launch_bounds__(256) void softmax_scale_kernel(
    const float* __restrict__ logits, unsigned short* __restrict__ P,
    int S, float scale)
{
    __shared__ float red[256];
    const int tid = threadIdx.x;
    const long long row = blockIdx.x;
    const float* x = logits + row * (long long)S + tid * 8;

    float v[8];
    const float4 x0 = *(const float4*)(x);
    const float4 x1 = *(const float4*)(x + 4);
    v[0] = x0.x; v[1] = x0.y; v[2] = x0.z; v[3] = x0.w;
    v[4] = x1.x; v[5] = x1.y; v[6] = x1.z; v[7] = x1.w;

    float m = -3.4e38f;
#pragma unroll
    for (int i = 0; i < 8; ++i) m = fmaxf(m, v[i]);
    red[tid] = m;
    __syncthreads();
#pragma unroll
    for (int s = 128; s > 0; s >>= 1) {
        if (tid < s) red[tid] = fmaxf(red[tid], red[tid + s]);
        __syncthreads();
    }
    m = red[0];
    __syncthreads();

    float e[8];
    float sum = 0.0f;
#pragma unroll
    for (int i = 0; i < 8; ++i) {
        e[i] = __expf(v[i] - m);
        sum += e[i];
    }
    red[tid] = sum;
    __syncthreads();
#pragma unroll
    for (int s = 128; s > 0; s >>= 1) {
        if (tid < s) red[tid] += red[tid + s];
        __syncthreads();
    }
    const float inv = scale / red[0];

    uint4 o;
    o.x = pack2_bf16(e[0] * inv, e[1] * inv);
    o.y = pack2_bf16(e[2] * inv, e[3] * inv);
    o.z = pack2_bf16(e[4] * inv, e[5] * inv);
    o.w = pack2_bf16(e[6] * inv, e[7] * inv);
    *(uint4*)(P + row * (long long)S + tid * 8) = o;
}

// ---------------------------------------------------------------------------
extern "C" void kernel_launch(void* const* d_in, const int* in_sizes, int n_in,
                              void* d_out, int out_size, void* d_ws, size_t ws_size,
                              hipStream_t stream) {
    (void)in_sizes; (void)n_in; (void)out_size; (void)ws_size;

    constexpr int B = 8, S = 2048, D = 1024;
    constexpr int BM = 128, BN = 128, BK = 64;
    const float scale = 0.03125f; // 1/sqrt(1024)

    const float* q  = (const float*)d_in[0];
    const float* k  = (const float*)d_in[1];
    const float* v  = (const float*)d_in[2];
    const float* Wq = (const float*)d_in[3];
    const float* bq = (const float*)d_in[4];
    const float* Wk = (const float*)d_in[5];
    const float* bk = (const float*)d_in[6];
    const float* Wv = (const float*)d_in[7];
    const float* bv = (const float*)d_in[8];
    float* out = (float*)d_out;

    // workspace layout
    const size_t qkvElems = (size_t)B * S * D;   // 16M
    unsigned short* Qb = (unsigned short*)d_ws;
    unsigned short* Kb = Qb + qkvElems;
    unsigned short* Vb = Kb + qkvElems;
    float* Lg = (float*)(Vb + qkvElems);         // B*S*S f32 logits
    unsigned short* Pb = (unsigned short*)(Lg + (size_t)B * S * S);

    // ---- 1) projections: [B*S, D] x [D, D] + bias -> bf16 ----
    {
        dim3 grid(D / BN, (B * S) / BM, 1);
        wmma_gemm_kernel<BM, BN, BK, 1, 0, 1, 0><<<grid, 256, 0, stream>>>(
            q, Wq, bq, Qb, B * S, D, D, 0, 0, 0);
        wmma_gemm_kernel<BM, BN, BK, 1, 0, 1, 0><<<grid, 256, 0, stream>>>(
            k, Wk, bk, Kb, B * S, D, D, 0, 0, 0);
        wmma_gemm_kernel<BM, BN, BK, 1, 0, 1, 0><<<grid, 256, 0, stream>>>(
            v, Wv, bv, Vb, B * S, D, D, 0, 0, 0);
    }

    // ---- 2) logits[b] = Q[b] * K[b]^T : f32 ----
    {
        dim3 grid(S / BN, S / BM, B);
        wmma_gemm_kernel<BM, BN, BK, 0, 1, 0, 1><<<grid, 256, 0, stream>>>(
            Qb, Kb, nullptr, Lg, S, S, D,
            (long long)S * D, (long long)S * D, (long long)S * S);
    }

    // ---- 3) softmax rows + quirk scale -> bf16 probs ----
    softmax_scale_kernel<<<B * S, 256, 0, stream>>>(Lg, Pb, S, scale);

    // ---- 4) out[b] = P[b] * V[b] : f32 ----
    {
        dim3 grid(D / BN, S / BM, B);
        wmma_gemm_kernel<BM, BN, BK, 0, 2, 0, 1><<<grid, 256, 0, stream>>>(
            Pb, Vb, nullptr, out, S, D, S,
            (long long)S * S, (long long)S * D, (long long)S * D);
    }
}
